// ActualBioInspiredModel_24730421690962
// MI455X (gfx1250) — matrix operand order repaired and checked by reading, not compiled, not run on hardware
//
#include <hip/hip_runtime.h>
#include <hip/hip_bf16.h>

// ---------------------------------------------------------------------------
// Problem constants (from reference)
// ---------------------------------------------------------------------------
#define BB     1024
#define D_IN   128
#define HID    64
#define ENH    84
#define NE     4
#define ED     16
#define VV     100000
#define NH     10

typedef __attribute__((ext_vector_type(16))) _Float16 v16h;
typedef __attribute__((ext_vector_type(8)))  float    v8f;

union Frag16 {
    v16h h;
    unsigned int u[8];
};

// ---------------------------------------------------------------------------
// Kernel 1: fused front-end. One block (64 threads) per batch row.
// proj -> phasor features -> expert gate softmax -> expert mix -> ctx (B,HID)
// ---------------------------------------------------------------------------
__global__ __launch_bounds__(64) void frontend_kernel(
    const float* __restrict__ x,   const float* __restrict__ W_in,
    const float* __restrict__ b_in,const float* __restrict__ Wg,
    const float* __restrict__ bg,  const float* __restrict__ We,
    const float* __restrict__ be,  const float* __restrict__ Wo,
    const float* __restrict__ bo,  float* __restrict__ ctx)
{
    __shared__ float enh[ENH];      // [0,64): proj  [64,74): cos  [74,84): sin
    __shared__ float gate[NE];
    __shared__ float eo[NE * ED];
    __shared__ float mixed[ED];
    __shared__ float xm;

    const int b   = blockIdx.x;
    const int tid = threadIdx.x;
    const float* xr = x + b * D_IN;

    // 1) input projection (one output column per thread)
    if (tid < HID) {
        float acc = b_in[tid];
        #pragma unroll 4
        for (int i = 0; i < D_IN; ++i)
            acc = fmaf(xr[i], W_in[i * HID + tid], acc);
        enh[tid] = acc;
    }
    __syncthreads();

    // 2) mean activation
    if (tid == 0) {
        float s = 0.f;
        for (int j = 0; j < HID; ++j) s += enh[j];
        xm = s * (1.0f / HID);
    }
    __syncthreads();

    // phasor temporal encoding
    if (tid < 2 * NH) {
        int   h  = (tid < NH) ? tid : tid - NH;
        float ph = xm * (7.0f * (float)(h + 1));
        enh[HID + tid] = (tid < NH) ? cosf(ph) : sinf(ph);
    }
    __syncthreads();

    // 3) gate logits
    if (tid < NE) {
        float acc = bg[tid];
        for (int i = 0; i < ENH; ++i)
            acc = fmaf(enh[i], Wg[i * NE + tid], acc);
        gate[tid] = acc;
    }
    __syncthreads();

    // softmax over 4 experts
    if (tid == 0) {
        float mx = gate[0];
        for (int e = 1; e < NE; ++e) mx = fmaxf(mx, gate[e]);
        float s = 0.f;
        for (int e = 0; e < NE; ++e) { gate[e] = __expf(gate[e] - mx); s += gate[e]; }
        float inv = 1.0f / s;
        for (int e = 0; e < NE; ++e) gate[e] *= inv;
    }
    __syncthreads();

    // expert outputs: tanh(enh @ We[e] + be[e]), one (e,o) per thread
    if (tid < NE * ED) {
        int e = tid / ED, o = tid % ED;
        float acc = be[e * ED + o];
        const float* Wep = We + (size_t)e * ENH * ED + o;
        for (int i = 0; i < ENH; ++i)
            acc = fmaf(enh[i], Wep[i * ED], acc);
        eo[tid] = tanhf(acc);
    }
    __syncthreads();

    // gate-weighted mix
    if (tid < ED) {
        float acc = 0.f;
        for (int e = 0; e < NE; ++e)
            acc = fmaf(gate[e], eo[e * ED + tid], acc);
        mixed[tid] = acc;
    }
    __syncthreads();

    // context vector
    if (tid < HID) {
        float acc = bo[tid];
        #pragma unroll
        for (int d = 0; d < ED; ++d)
            acc = fmaf(mixed[d], Wo[d * HID + tid], acc);
        ctx[b * HID + tid] = acc;
    }
}

// ---------------------------------------------------------------------------
// Kernel 2: spike/top-k gain logic on ctx row 0. 64 elements -> serial thread.
// argsort(|ctx[0]|) stable ascending, take last 20, scatter decayed weights,
// 5 winners (lowest-index ties), gain=2 where winner potential >= THETA.
// ---------------------------------------------------------------------------
__global__ void gains_kernel(const float* __restrict__ ctx, float* __restrict__ gains)
{
    if (threadIdx.x != 0 || blockIdx.x != 0) return;

    float v[HID]; int idx[HID];
    for (int j = 0; j < HID; ++j) { v[j] = fabsf(ctx[j]); idx[j] = j; }

    // stable insertion sort ascending (matches jnp.argsort stable)
    for (int i = 1; i < HID; ++i) {
        float kv = v[i]; int ki = idx[i]; int j = i - 1;
        while (j >= 0 && v[j] > kv) { v[j + 1] = v[j]; idx[j + 1] = idx[j]; --j; }
        v[j + 1] = kv; idx[j + 1] = ki;
    }

    float pot[HID];
    for (int j = 0; j < HID; ++j) pot[j] = 0.f;
    for (int t = 0; t < 20; ++t) {
        int id = idx[HID - 20 + t] % VV;        // indices < 64, %V is identity
        pot[id] += powf(0.7f, (float)(19 - t)); // DECAY^(19-t)
    }

    float g[HID];
    for (int j = 0; j < HID; ++j) g[j] = 1.f;
    for (int it = 0; it < 5; ++it) {            // K_WIN = 5 winners
        int best = -1; float bv = -1.f;
        for (int j = 0; j < HID; ++j)
            if (pot[j] > bv) { bv = pot[j]; best = j; }
        g[best] = (bv >= 1.0f) ? 2.0f : 1.0f;   // THETA = 1.0
        pot[best] = -2.f;
    }
    for (int j = 0; j < HID; ++j) gains[j] = g[j];
}

// ---------------------------------------------------------------------------
// Kernel 3: attended = ctx * gains[:HID], converted to f16 (A operand)
// ---------------------------------------------------------------------------
__global__ __launch_bounds__(256) void att_convert_kernel(
    const float* __restrict__ ctx, const float* __restrict__ gains,
    _Float16* __restrict__ att)
{
    int i = blockIdx.x * 256 + threadIdx.x;
    if (i < BB * HID)
        att[i] = (_Float16)(ctx[i] * gains[i & (HID - 1)]);
}

// ---------------------------------------------------------------------------
// Kernel 4: W_out (K=64 x V=100000 f32, row-major) -> f16 transposed N-major
// (V x 64). LDS tiled so both global reads and writes are coalesced.
// ---------------------------------------------------------------------------
__global__ __launch_bounds__(256) void wout_convert_kernel(
    const float* __restrict__ W, _Float16* __restrict__ BT)
{
    __shared__ float tile[64][33];          // +1 pad: bank-conflict free
    const int n0  = blockIdx.x * 32;
    const int tid = threadIdx.x;
    const int ni  = tid & 31, kq = tid >> 5;            // 8 k-rows per pass
    #pragma unroll
    for (int kk = 0; kk < 8; ++kk) {
        int k = kk * 8 + kq;
        tile[k][ni] = W[(size_t)k * VV + n0 + ni];      // coalesced over n
    }
    __syncthreads();
    const int k = tid & 63, nq = tid >> 6;              // 4 n-rows per pass
    #pragma unroll
    for (int jj = 0; jj < 8; ++jj) {
        int nn = jj * 4 + nq;
        BT[(size_t)(n0 + nn) * 64 + k] = (_Float16)tile[k][nn]; // 128B rows
    }
}

// ---------------------------------------------------------------------------
// Kernel 5: the bandwidth-bound GEMM  out[1024 x 100000] = A(f16) @ BT(f16)^T
// + bias, via v_wmma_f32_16x16x32_f16 (K=64 -> 2 k-steps). One wave per
// 16x16 tile, 8 consecutive N-tiles per wave reusing the A fragments.
// A/B fragments use contiguous k (k = 16*half + s) for both; f32 D 16x16 per CDNA5 ISA 7.12.2.
// ---------------------------------------------------------------------------
#define GEMM_WAVES 4
#define GEMM_NT    8
#define N_TILES    (VV / 16)     /* 6250, exact */

__global__ __launch_bounds__(32 * GEMM_WAVES) void wmma_gemm_kernel(
    const _Float16* __restrict__ A,    // 1024 x 64, row-major
    const _Float16* __restrict__ BT,   // 100000 x 64, n-major
    const float*    __restrict__ bias, // 100000
    float*          __restrict__ out)  // 1024 x 100000
{
    const int lane = threadIdx.x & 31;
    const int wave = threadIdx.x >> 5;
    const int half = (lane >> 4) & 1;   // 0: lanes 0-15, 1: lanes 16-31
    const int ml   = lane & 15;

    const int tile_m  = blockIdx.y;
    const int tn_base = (blockIdx.x * GEMM_WAVES + wave) * GEMM_NT;

    // --- A fragments for both k-steps (kept in registers across N tiles) ---
    Frag16 a[2];
    const _Float16* Arow = A + (size_t)(tile_m * 16 + ml) * HID;
    #pragma unroll
    for (int ks = 0; ks < 2; ++ks) {
        const int kb = ks * 32 + half * 16;
        #pragma unroll
        for (int v = 0; v < 8; ++v) {
            const int k = kb + 2 * v;
            a[ks].u[v] = *(const unsigned int*)(Arow + k);  // (k even -> dword aligned)
        }
    }

    for (int t = 0; t < GEMM_NT; ++t) {
        const int tn = tn_base + t;
        if (tn >= N_TILES) break;              // wave-uniform; EXEC stays full
        const int n = tn * 16 + ml;
        const _Float16* Brow = BT + (size_t)n * HID;

        v8f c = {};
        #pragma unroll
        for (int ks = 0; ks < 2; ++ks) {
            Frag16 bfr;
            const int kb = ks * 32 + half * 16;
            #pragma unroll
            for (int v = 0; v < 8; ++v)
                bfr.u[v] = *(const unsigned int*)(Brow + kb + 2 * v);
            c = __builtin_amdgcn_wmma_f32_16x16x32_f16(
                    /*neg_a=*/false, a[ks].h, /*neg_b=*/false, bfr.h,
                    /*c_mod=*/(short)0, c, /*reuse_a=*/false, /*reuse_b=*/false);
        }

        const float bo = bias[n];
        const int mbase = tile_m * 16 + half * 8;
        #pragma unroll
        for (int v = 0; v < 8; ++v)
            out[(size_t)(mbase + v) * VV + n] = c[v] + bo;
    }
}

// ---------------------------------------------------------------------------
// Host-side launcher
// ---------------------------------------------------------------------------
extern "C" void kernel_launch(void* const* d_in, const int* in_sizes, int n_in,
                              void* d_out, int out_size, void* d_ws, size_t ws_size,
                              hipStream_t stream)
{
    const float* x     = (const float*)d_in[0];
    const float* W_in  = (const float*)d_in[1];
    const float* b_in  = (const float*)d_in[2];
    const float* Wg    = (const float*)d_in[3];
    const float* bg    = (const float*)d_in[4];
    const float* We    = (const float*)d_in[5];
    const float* be    = (const float*)d_in[6];
    const float* Wo    = (const float*)d_in[7];
    const float* bo    = (const float*)d_in[8];
    const float* W_out = (const float*)d_in[9];
    const float* b_out = (const float*)d_in[10];
    float*       out   = (float*)d_out;

    // workspace layout (256B aligned offsets)
    char* ws = (char*)d_ws;
    float*     ctx   = (float*)(ws);                      // 1024*64*4   = 262144
    float*     gains = (float*)(ws + 262144);             // 64*4        (pad 256)
    _Float16*  attA  = (_Float16*)(ws + 262400);          // 1024*64*2   = 131072
    _Float16*  BT    = (_Float16*)(ws + 393472);          // 100000*64*2 = 12.8MB

    // 1) fused front-end -> ctx
    frontend_kernel<<<BB, 64, 0, stream>>>(x, W_in, b_in, Wg, bg, We, be, Wo, bo, ctx);

    // 2) spike/top-k gains from ctx row 0
    gains_kernel<<<1, 32, 0, stream>>>(ctx, gains);

    // 3) attended (f16 A operand)
    att_convert_kernel<<<(BB * HID + 255) / 256, 256, 0, stream>>>(ctx, gains, attA);

    // 4) W_out -> f16 transposed (n-major) B operand
    wout_convert_kernel<<<VV / 32, 256, 0, stream>>>(W_out, BT);

    // 5) WMMA GEMM + bias -> logits
    dim3 grid((N_TILES + GEMM_WAVES * GEMM_NT - 1) / (GEMM_WAVES * GEMM_NT), BB / 16);
    wmma_gemm_kernel<<<grid, 32 * GEMM_WAVES, 0, stream>>>(attA, BT, b_out, out);
}
